// IEGMN_Bidirectional_Layer_34093450395756
// MI455X (gfx1250) — matrix operand order repaired and checked by reading, not compiled, not run on hardware
//
#include <hip/hip_runtime.h>

#define NNODES 10000
#define NEDGES 320000
#define DIM    64
#define NPAD   10016            // NNODES rounded up to 32
#define EINK   160              // 2*D + 15 = 143, padded to mult. of 32
#define NEGS   0.01f
#define NQT    (NNODES / 16)    // 625 node row-tiles
#define NET    (NEDGES / 16)    // 20000 edge row-tiles

typedef __attribute__((ext_vector_type(16))) __bf16 bf16x16;
typedef __attribute__((ext_vector_type(8)))  __bf16 bf16x8;
typedef __attribute__((ext_vector_type(8)))  float  f32x8;
typedef __attribute__((ext_vector_type(4)))  int    v4i;

// ------------------------------------------------------- async LDS copies

__device__ __forceinline__ void async_copy16(void* lds, const void* g) {
#if __has_builtin(__builtin_amdgcn_global_load_async_to_lds_b128)
  // signature (probe-confirmed): (int4* lds_dst, int4* global_src, Ii, Ii)
  __builtin_amdgcn_global_load_async_to_lds_b128((v4i*)lds, (v4i*)g, 0, 0);
#else
  unsigned loff = (unsigned)(size_t)lds;   // generic LDS addr: low 32 = offset
  asm volatile("global_load_async_to_lds_b128 %0, %1, off"
               :: "v"(loff), "v"(g) : "memory");
#endif
}

__device__ __forceinline__ void wait_async0() {
#if __has_builtin(__builtin_amdgcn_s_wait_asynccnt)
  __builtin_amdgcn_s_wait_asynccnt(0);
#else
  asm volatile("s_wait_asynccnt 0" ::: "memory");
#endif
}

__device__ __forceinline__ void wait_ds0() {
  asm volatile("s_wait_dscnt 0" ::: "memory");
}

// ---------------------------------------------------------------- fragments

__device__ __forceinline__ f32x8 wmma_bf16(bf16x16 a, bf16x16 b, f32x8 c) {
  return __builtin_amdgcn_wmma_f32_16x16x32_bf16(false, a, false, b,
                                                 (short)0, c, false, false);
}

// A fragment (16x32 bf16, row-major source, row m = lane&15):
//  a[0..7]  = K: koff + (lane<16 ? 0 : 16) + 0..7
//  a[8..15] = K: koff + (lane<16 ? 8 : 24) + 0..7
__device__ __forceinline__ bf16x16 load_afrag(const __bf16* A, int lda,
                                              int koff, int lane) {
  const __bf16* p = A + (lane & 15) * lda + koff + ((lane >> 4) << 4);
  bf16x8 lo = *(const bf16x8*)p;
  bf16x8 hi = *(const bf16x8*)(p + 8);
  return __builtin_shufflevector(lo, hi, 0,1,2,3,4,5,6,7,
                                 8,9,10,11,12,13,14,15);
}

// B fragment (32x16 bf16) from TRANSPOSED operand Bt[outCol][K]:
//  col n = n0 + (lane&15); b[j] = Bt[n][koff + j + (lane<16 ? 0 : 16)]
__device__ __forceinline__ bf16x16 load_bfrag(const __bf16* Bt, int ldk,
                                              int n0, int koff, int lane) {
  return *(const bf16x16*)(Bt + (size_t)(n0 + (lane & 15)) * ldk +
                           koff + ((lane >> 4) << 4));
}

// ---------------------------------------------------------------- reductions
// D-fragment: row m = r + 8*(lane>=16), col n = lane&15  -> half-wave reduce.

__device__ __forceinline__ float hmax16(float v) {
#pragma unroll
  for (int d = 1; d < 16; d <<= 1) v = fmaxf(v, __shfl_xor(v, d, 16));
  return v;
}
__device__ __forceinline__ float hsum16(float v) {
#pragma unroll
  for (int d = 1; d < 16; d <<= 1) v += __shfl_xor(v, d, 16);
  return v;
}

// 16 rows x 64 cols output tile: c[t] = A(16xK) * Bt (+ bias)
__device__ __forceinline__ void gemm16(const __bf16* A, int lda,
                                       const __bf16* Bt, int K,
                                       const float* bias, int lane, f32x8* c) {
#pragma unroll
  for (int t = 0; t < 4; ++t) {
    f32x8 acc = {0.f,0.f,0.f,0.f,0.f,0.f,0.f,0.f};
    for (int k = 0; k < K; k += 32) {
      bf16x16 a = load_afrag(A, lda, k, lane);
      bf16x16 b = load_bfrag(Bt, K, t * 16, k, lane);
      acc = wmma_bf16(a, b, acc);
    }
    if (bias) {                       // folded away at nullptr call sites
      float bv = bias[t * 16 + (lane & 15)];
#pragma unroll
      for (int r = 0; r < 8; ++r) acc[r] += bv;
    }
    c[t] = acc;
  }
}

// LayerNorm over 64 features (rows spread across half-wave), optional lrelu
__device__ __forceinline__ void lnorm16(f32x8* c, const float* g,
                                        const float* bt, int lane, bool lrelu) {
  int n = lane & 15;
#pragma unroll
  for (int r = 0; r < 8; ++r) {
    float s = 0.f, s2 = 0.f;
#pragma unroll
    for (int t = 0; t < 4; ++t) { float x = c[t][r]; s += x; s2 += x * x; }
    s = hsum16(s); s2 = hsum16(s2);
    float mean = s * (1.f / 64.f);
    float var  = s2 * (1.f / 64.f) - mean * mean;
    float inv  = rsqrtf(var + 1e-5f);
#pragma unroll
    for (int t = 0; t < 4; ++t) {
      float x = (c[t][r] - mean) * inv * g[t * 16 + n] + bt[t * 16 + n];
      if (lrelu) x = x >= 0.f ? x : NEGS * x;
      c[t][r] = x;
    }
  }
}

// store D-fragments as bf16 row-major [16][ld] into LDS (A-staging)
__device__ __forceinline__ void stage16(__bf16* S, int ld, const f32x8* c,
                                        int lane) {
  int n = lane & 15, hi = (lane >> 4) << 3;
#pragma unroll
  for (int t = 0; t < 4; ++t)
#pragma unroll
    for (int r = 0; r < 8; ++r)
      S[(r + hi) * ld + t * 16 + n] = (__bf16)c[t][r];
}

// ---------------------------------------------------------------- prep

__global__ void k_f2bf(const float* x, __bf16* y, int total) {
  for (int i = blockIdx.x * blockDim.x + threadIdx.x; i < total;
       i += gridDim.x * blockDim.x)
    y[i] = (__bf16)x[i];
}

// W [K][Nout] fp32 -> Wt [Nout][Kpad] bf16 (zero pad K)
__global__ void k_wT(const float* W, __bf16* Wt, int K, int Nout, int Kpad) {
  int total = Nout * Kpad;
  for (int i = blockIdx.x * blockDim.x + threadIdx.x; i < total;
       i += gridDim.x * blockDim.x) {
    int n = i / Kpad, k = i % Kpad;
    Wt[i] = (k < K) ? (__bf16)W[k * Nout + n] : (__bf16)0.f;
  }
}

// ---------------------------------------------------------------- QKV
__global__ void __launch_bounds__(256)
k_qkv(const __bf16* h_bf, const __bf16* WQt, const __bf16* WKt,
      const __bf16* WVt, __bf16* q_bf, __bf16* k_bf, __bf16* vT) {
  int lane = threadIdx.x & 31, wave = threadIdx.x >> 5;
  int tile = blockIdx.x * 8 + wave;
  if (tile >= NQT) return;
  const __bf16* A = h_bf + (size_t)tile * 16 * DIM;
  int n = lane & 15, hi = (lane >> 4) << 3;
  f32x8 c[4];

  gemm16(A, DIM, WQt, DIM, nullptr, lane, c);
#pragma unroll
  for (int t = 0; t < 4; ++t)
#pragma unroll
    for (int r = 0; r < 8; ++r) {
      float x = c[t][r]; x = x >= 0.f ? x : NEGS * x;   // lrelu(q)
      q_bf[(size_t)(tile * 16 + r + hi) * DIM + t * 16 + n] = (__bf16)x;
    }
  gemm16(A, DIM, WKt, DIM, nullptr, lane, c);
#pragma unroll
  for (int t = 0; t < 4; ++t)
#pragma unroll
    for (int r = 0; r < 8; ++r) {
      float x = c[t][r]; x = x >= 0.f ? x : NEGS * x;   // lrelu(k)
      k_bf[(size_t)(tile * 16 + r + hi) * DIM + t * 16 + n] = (__bf16)x;
    }
  gemm16(A, DIM, WVt, DIM, nullptr, lane, c);
#pragma unroll
  for (int t = 0; t < 4; ++t)
#pragma unroll
    for (int r = 0; r < 8; ++r)                         // V stored transposed
      vT[(size_t)(t * 16 + n) * NPAD + tile * 16 + r + hi] = (__bf16)c[t][r];
}

// ------------------------------------------------------------- attention
// Flash-style online softmax. All 8 waves of a block walk the same key
// sequence; K/V tiles are double-buffered in LDS and filled with
// global_load_async_to_lds_b128 (ASYNCcnt), overlapping WMMA compute.
__global__ void __launch_bounds__(256)
k_attn(const __bf16* q_bf, const __bf16* k_bf, const __bf16* vT,
       const float* mask, float* cross) {
  __shared__ __align__(32) __bf16 Kt[2][32 * 64];   // [key][d]    4KB/buf
  __shared__ __align__(32) __bf16 Vt[2][64 * 32];   // [dout][key] 4KB/buf
  __shared__ __align__(32) __bf16 Pbuf[8][16 * 32];
  int tid = threadIdx.x;
  int lane = tid & 31, wave = tid >> 5;
  int tile0 = blockIdx.x * 8 + wave;
  int tile  = tile0 < NQT ? tile0 : NQT - 1;        // clamp; store guarded
  __bf16* P = Pbuf[wave];
  int n = lane & 15, hi = (lane >> 4) << 3;
  int qbase = tile * 16;

  // per-thread async chunk assignment (16B per op, whole block covers tile)
  int krow = tid >> 3, kch = tid & 7;               // K: 32 rows x 8 chunks
  int vrow = tid >> 2, vch = tid & 3;               // V: 64 rows x 4 chunks

  bf16x16 aq0 = load_afrag(q_bf + (size_t)qbase * DIM, DIM, 0, lane);
  bf16x16 aq1 = load_afrag(q_bf + (size_t)qbase * DIM, DIM, 32, lane);

  f32x8 o[4];
#pragma unroll
  for (int t = 0; t < 4; ++t) o[t] = (f32x8){0.f,0.f,0.f,0.f,0.f,0.f,0.f,0.f};
  float mrow[8], lrow[8];
#pragma unroll
  for (int r = 0; r < 8; ++r) { mrow[r] = -3.0e38f; lrow[r] = 0.f; }

  const int NKT = (NNODES + 31) / 32;   // 313 key tiles (pad rows zeroed)
  // prologue: stage tile 0
  async_copy16(&Kt[0][krow * 64 + kch * 8],
               k_bf + (size_t)krow * DIM + kch * 8);
  async_copy16(&Vt[0][vrow * 32 + vch * 8],
               vT + (size_t)vrow * NPAD + vch * 8);

  for (int kb = 0; kb < NKT; ++kb) {
    int kbase = kb * 32;
    wait_async0();            // own copies for tile kb landed
    __syncthreads();          // everyone's landed; prev readers done (no WAR)
    if (kb + 1 < NKT) {       // overlap next tile's fill with compute
      int nb = kbase + 32;
      async_copy16(&Kt[(kb + 1) & 1][krow * 64 + kch * 8],
                   k_bf + (size_t)(nb + krow) * DIM + kch * 8);
      async_copy16(&Vt[(kb + 1) & 1][vrow * 32 + vch * 8],
                   vT + (size_t)vrow * NPAD + nb + vch * 8);
    }
    const __bf16* KT = Kt[kb & 1];
    const __bf16* VT = Vt[kb & 1];

    f32x8 s0 = {0.f,0.f,0.f,0.f,0.f,0.f,0.f,0.f};
    f32x8 s1 = s0;
    s0 = wmma_bf16(aq0, load_bfrag(KT, 64, 0,  0,  lane), s0);
    s0 = wmma_bf16(aq1, load_bfrag(KT, 64, 0,  32, lane), s0);
    s1 = wmma_bf16(aq0, load_bfrag(KT, 64, 16, 0,  lane), s1);
    s1 = wmma_bf16(aq1, load_bfrag(KT, 64, 16, 32, lane), s1);

    int keyA = kbase + n, keyB = keyA + 16;
#pragma unroll
    for (int r = 0; r < 8; ++r) {
      int qrow = qbase + r + hi;
      float t0, t1;
      if (keyA < NNODES) {
        float mv = __builtin_nontemporal_load(mask + (size_t)qrow * NNODES + keyA);
        t0 = mv * s0[r] - 1000.f * (1.f - mv);
      } else t0 = -1e30f;
      if (keyB < NNODES) {
        float mv = __builtin_nontemporal_load(mask + (size_t)qrow * NNODES + keyB);
        t1 = mv * s1[r] - 1000.f * (1.f - mv);
      } else t1 = -1e30f;
      float mx    = hmax16(fmaxf(t0, t1));
      float m_new = fmaxf(mrow[r], mx);
      float p0 = __expf(t0 - m_new), p1 = __expf(t1 - m_new);
      float ps   = hsum16(p0 + p1);
      float corr = __expf(mrow[r] - m_new);
      lrow[r] = lrow[r] * corr + ps;
      mrow[r] = m_new;
#pragma unroll
      for (int t = 0; t < 4; ++t) o[t][r] *= corr;
      P[(r + hi) * 32 + n]      = (__bf16)p0;
      P[(r + hi) * 32 + 16 + n] = (__bf16)p1;
    }
    wait_ds0();
    bf16x16 ap = load_afrag(P, 32, 0, lane);
#pragma unroll
    for (int t = 0; t < 4; ++t)
      o[t] = wmma_bf16(ap, load_bfrag(VT, 32, t * 16, 0, lane), o[t]);
  }
  if (tile0 < NQT) {
#pragma unroll
    for (int r = 0; r < 8; ++r) {
      float invl = 1.f / lrow[r];
      int qrow = qbase + r + hi;
#pragma unroll
      for (int t = 0; t < 4; ++t)
        cross[(size_t)qrow * DIM + t * 16 + n] = o[t][r] * invl;
    }
  }
}

// ---------------------------------------------------------------- edges
__global__ void __launch_bounds__(128)
k_edge(const int* src, const int* dst, const float* coords,
       const __bf16* h_bf,
       const __bf16* We1t, const float* be1, const float* ge1, const float* bte1,
       const __bf16* We2t, const float* be2, const float* ge2, const float* bte2,
       const __bf16* Wc1t, const float* bc1, const float* Wc2, const float* bc2,
       float* aggr, float* xu, float* cnt) {
  __shared__ __align__(32) __bf16 Ein[4][16 * EINK];
  int lane = threadIdx.x & 31, wave = threadIdx.x >> 5;
  int tile = blockIdx.x * 4 + wave;
  if (tile >= NET) return;
  __bf16* A = Ein[wave];
  int ebase = tile * 16;
  int n = lane & 15, hi = (lane >> 4) << 3;

  // stage e_in = [h[src] | h[dst] | rbf | zeros] as 16x160 bf16:
  // lanes 0-7 async-copy h[src] (8x16B), lanes 8-15 h[dst], lanes 16-31 rbf.
  for (int e = 0; e < 16; ++e) {
    int eid = ebase + e;
    int s = src[eid], d = dst[eid];
    if (lane < 8) {
      async_copy16(&A[e * EINK + lane * 8],
                   h_bf + (size_t)s * DIM + lane * 8);
    } else if (lane < 16) {
      async_copy16(&A[e * EINK + 64 + (lane - 8) * 8],
                   h_bf + (size_t)d * DIM + (lane - 8) * 8);
    } else {
      int j = lane - 16;                       // 0..15
      float rx = coords[s * 3 + 0] - coords[d * 3 + 0];
      float ry = coords[s * 3 + 1] - coords[d * 3 + 1];
      float rz = coords[s * 3 + 2] - coords[d * 3 + 2];
      float mag = rx * rx + ry * ry + rz * rz;
      float v = (j < 15) ? __expf(-mag / __powf(1.5f, (float)j)) : 0.f;
      A[e * EINK + 128 + j] = (__bf16)v;
      A[e * EINK + 144 + j] = (__bf16)0.f;     // K padding
    }
  }
  wait_async0();
  wait_ds0();

  f32x8 c[4];
  gemm16(A, EINK, We1t, EINK, be1, lane, c);   // Linear1
  lnorm16(c, ge1, bte1, lane, true);           // LN + lrelu
  stage16(A, 64, c, lane);                     // reuse LDS as 16x64
  wait_ds0();
  gemm16(A, 64, We2t, 64, be2, lane, c);       // Linear2
  lnorm16(c, ge2, bte2, lane, false);          // LN -> msg

  f32x8 msg_[4];
#pragma unroll
  for (int t = 0; t < 4; ++t) msg_[t] = c[t];
  stage16(A, 64, c, lane);                     // msg bf16 for coef GEMM
  wait_ds0();

  f32x8 c3[4];
  gemm16(A, 64, Wc1t, 64, bc1, lane, c3);      // coef hidden
  float bc2v = bc2[0];
  float coef[8];
#pragma unroll
  for (int r = 0; r < 8; ++r) {
    float p = 0.f;
#pragma unroll
    for (int t = 0; t < 4; ++t) {
      float x = c3[t][r]; x = x >= 0.f ? x : NEGS * x;   // lrelu
      p += x * Wc2[t * 16 + n];
    }
    coef[r] = hsum16(p) + bc2v;
  }
  // scatter: segment sums to dst (mean finalized in node kernel via cnt)
#pragma unroll
  for (int r = 0; r < 8; ++r) {
    int eid = ebase + r + hi;
    int d = dst[eid];
#pragma unroll
    for (int t = 0; t < 4; ++t)
      atomicAdd(&aggr[(size_t)d * DIM + t * 16 + n], msg_[t][r]);
    if (n == 0) {
      int s = src[eid];
      float rx = coords[s * 3 + 0] - coords[d * 3 + 0];
      float ry = coords[s * 3 + 1] - coords[d * 3 + 1];
      float rz = coords[s * 3 + 2] - coords[d * 3 + 2];
      atomicAdd(&xu[d * 3 + 0], rx * coef[r]);
      atomicAdd(&xu[d * 3 + 1], ry * coef[r]);
      atomicAdd(&xu[d * 3 + 2], rz * coef[r]);
      atomicAdd(&cnt[d], 1.f);
    }
  }
}

// ---------------------------------------------------------------- nodes
__global__ void __launch_bounds__(128)
k_node(const float* h, const float* origh, const float* coords,
       const float* origc, const float* cross, const float* aggr,
       const float* xu, const float* cnt,
       const __bf16* Wn1t, const float* bn1, const float* gn1, const float* btn1,
       const __bf16* Wn2t, const float* bn2, const float* gn2, const float* btn2,
       float* out) {
  __shared__ __align__(32) __bf16 Cat[4][16 * 256];
  int lane = threadIdx.x & 31, wave = threadIdx.x >> 5;
  int tile = blockIdx.x * 4 + wave;
  if (tile >= NQT) return;
  __bf16* A = Cat[wave];
  int base = tile * 16;
  int n = lane & 15, hi = (lane >> 4) << 3;

  for (int m = 0; m < 16; ++m) {
    int node = base + m;
    float invc = 1.f / fmaxf(cnt[node], 1.f);
    for (int i = lane; i < 256; i += 32) {
      float v;
      if (i < 64)       v = h[(size_t)node * DIM + i];
      else if (i < 128) v = aggr[(size_t)node * DIM + (i - 64)] * invc;
      else if (i < 192) v = cross[(size_t)node * DIM + (i - 128)];
      else              v = origh[(size_t)node * DIM + (i - 192)];
      A[m * 256 + i] = (__bf16)v;
    }
  }
  wait_ds0();

  f32x8 c[4];
  gemm16(A, 256, Wn1t, 256, bn1, lane, c);
  lnorm16(c, gn1, btn1, lane, true);
  stage16(A, 64, c, lane);
  wait_ds0();
  gemm16(A, 64, Wn2t, 64, bn2, lane, c);
  lnorm16(c, gn2, btn2, lane, false);

  float* node_out = out + (size_t)NNODES * 3;
#pragma unroll
  for (int r = 0; r < 8; ++r) {
    int node = base + r + hi;
#pragma unroll
    for (int t = 0; t < 4; ++t) {
      float hv = h[(size_t)node * DIM + t * 16 + n];
      node_out[(size_t)node * DIM + t * 16 + n] = 0.5f * c[t][r] + 0.5f * hv;
    }
    if (n == 0) {
      float invc = 1.f / fmaxf(cnt[node], 1.f);
#pragma unroll
      for (int j = 0; j < 3; ++j)
        out[node * 3 + j] = 0.25f * origc[node * 3 + j] +
                            0.75f * coords[node * 3 + j] +
                            xu[node * 3 + j] * invc;
    }
  }
}

// ---------------------------------------------------------------- launch
extern "C" void kernel_launch(void* const* d_in, const int* in_sizes, int n_in,
                              void* d_out, int out_size, void* d_ws,
                              size_t ws_size, hipStream_t stream) {
  (void)in_sizes; (void)n_in; (void)out_size; (void)ws_size;
  const int*   src    = (const int*)d_in[0];
  const int*   dst    = (const int*)d_in[1];
  const float* coords = (const float*)d_in[2];
  const float* h      = (const float*)d_in[3];
  const float* origh  = (const float*)d_in[4];
  const float* origc  = (const float*)d_in[5];
  const float* mask   = (const float*)d_in[6];
  const float* We1 = (const float*)d_in[7],  *be1 = (const float*)d_in[8];
  const float* ge1 = (const float*)d_in[9],  *bte1= (const float*)d_in[10];
  const float* We2 = (const float*)d_in[11], *be2 = (const float*)d_in[12];
  const float* ge2 = (const float*)d_in[13], *bte2= (const float*)d_in[14];
  const float* WQ  = (const float*)d_in[15], *WK  = (const float*)d_in[16];
  const float* WV  = (const float*)d_in[17];
  const float* Wn1 = (const float*)d_in[18], *bn1 = (const float*)d_in[19];
  const float* gn1 = (const float*)d_in[20], *btn1= (const float*)d_in[21];
  const float* Wn2 = (const float*)d_in[22], *bn2 = (const float*)d_in[23];
  const float* gn2 = (const float*)d_in[24], *btn2= (const float*)d_in[25];
  const float* Wc1 = (const float*)d_in[26], *bc1 = (const float*)d_in[27];
  const float* Wc2 = (const float*)d_in[28], *bc2 = (const float*)d_in[29];

  char* ws = (char*)d_ws;
  size_t off = 0;
  auto take = [&](size_t bytes) {
    char* p = ws + off; off += (bytes + 255) & ~(size_t)255; return p;
  };
  __bf16* h_bf = (__bf16*)take((size_t)NNODES * DIM * 2);
  __bf16* q_bf = (__bf16*)take((size_t)NPAD * DIM * 2);
  __bf16* k_bf = (__bf16*)take((size_t)NPAD * DIM * 2);
  __bf16* vT   = (__bf16*)take((size_t)DIM * NPAD * 2);
  float*  cross= (float*) take((size_t)NNODES * DIM * 4);
  float*  aggr = (float*) take((size_t)NNODES * DIM * 4);
  float*  xu   = (float*) take((size_t)NNODES * 3 * 4);
  float*  cnt  = (float*) take((size_t)NNODES * 4);
  __bf16* We1t = (__bf16*)take(64 * EINK * 2);
  __bf16* We2t = (__bf16*)take(64 * 64 * 2);
  __bf16* WQt  = (__bf16*)take(64 * 64 * 2);
  __bf16* WKt  = (__bf16*)take(64 * 64 * 2);
  __bf16* WVt  = (__bf16*)take(64 * 64 * 2);
  __bf16* Wn1t = (__bf16*)take(64 * 256 * 2);
  __bf16* Wn2t = (__bf16*)take(64 * 64 * 2);
  __bf16* Wc1t = (__bf16*)take(64 * 64 * 2);

  (void)hipMemsetAsync(q_bf, 0, (size_t)NPAD * DIM * 2, stream);
  (void)hipMemsetAsync(k_bf, 0, (size_t)NPAD * DIM * 2, stream);
  (void)hipMemsetAsync(vT,   0, (size_t)DIM * NPAD * 2, stream);
  (void)hipMemsetAsync(aggr, 0, (size_t)NNODES * DIM * 4, stream);
  (void)hipMemsetAsync(xu,   0, (size_t)NNODES * 3 * 4, stream);
  (void)hipMemsetAsync(cnt,  0, (size_t)NNODES * 4, stream);

  k_f2bf<<<2500, 256, 0, stream>>>(h, h_bf, NNODES * DIM);
  auto WT = [&](const float* W, __bf16* Wt, int K, int Nout, int Kpad) {
    int tot = Nout * Kpad;
    k_wT<<<(tot + 255) / 256, 256, 0, stream>>>(W, Wt, K, Nout, Kpad);
  };
  WT(We1, We1t, 143, 64, EINK);
  WT(We2, We2t, 64, 64, 64);
  WT(WQ,  WQt,  64, 64, 64);
  WT(WK,  WKt,  64, 64, 64);
  WT(WV,  WVt,  64, 64, 64);
  WT(Wn1, Wn1t, 256, 64, 256);
  WT(Wn2, Wn2t, 64, 64, 64);
  WT(Wc1, Wc1t, 64, 64, 64);

  k_qkv <<<(NQT + 7) / 8, 256, 0, stream>>>(h_bf, WQt, WKt, WVt,
                                            q_bf, k_bf, vT);
  k_attn<<<(NQT + 7) / 8, 256, 0, stream>>>(q_bf, k_bf, vT, mask, cross);
  k_edge<<<NET / 4, 128, 0, stream>>>(src, dst, coords, h_bf,
                                      We1t, be1, ge1, bte1,
                                      We2t, be2, ge2, bte2,
                                      Wc1t, bc1, Wc2, bc2,
                                      aggr, xu, cnt);
  k_node<<<(NQT + 3) / 4, 128, 0, stream>>>(h, origh, coords, origc,
                                            cross, aggr, xu, cnt,
                                            Wn1t, bn1, gn1, btn1,
                                            Wn2t, bn2, gn2, btn2,
                                            (float*)d_out);
}
